// LongNetLM_49916109914348
// MI455X (gfx1250) — matrix operand order repaired and checked by reading, not compiled, not run on hardware
//
#include <hip/hip_runtime.h>

// ---------------------------------------------------------------------------
// LongNet (dilated-attention encoder/decoder LM) forward for MI455X (gfx1250).
// All matrix math via v_wmma_f32_16x16x32_bf16 (wave32 WMMA):
//  - tiled GEMM (projections, FFN, vocab head), fp32->bf16 staged in LDS with
//    the B operand stored TRANSPOSED; all WMMA fragments load as b128 pairs.
//  - fused flash attention over packed dilated segments (sp==1024 for all
//    three configs); K/V tiles staged with GLOBAL_LOAD_ASYNC_TO_LDS_B128
//    (ASYNCcnt path, no VGPR round-trip); V packed transposed so PV
//    fragments are contiguous.
// 192MB L2 absorbs fp32 weight re-reads; at 23.3TB/s HBM the ~420 GFLOP of
// matmul dominates, so bf16 WMMA with fp32 accumulate is the right path.
// ---------------------------------------------------------------------------

#define DMODEL 768
#define NHEADS 12
#define HDIM   64
#define DFF    3072
#define VOCABN 16000
#define SEQN   4096
#define SP     1024   // packed sub-sequence length (s / r) for every config

typedef __bf16 bf16_t;
typedef __attribute__((ext_vector_type(16))) __bf16        v16bf;
typedef __attribute__((ext_vector_type(8)))  __bf16        v8bf;
typedef __attribute__((ext_vector_type(8)))  float         v8f;
typedef __attribute__((ext_vector_type(8)))  unsigned int  v8u;

__device__ __forceinline__ bf16_t f2bf(float f) {
  return (bf16_t)f;   // native v_cvt (RNE) on gfx1250
}

__device__ __forceinline__ v8f wmma_bf16(v16bf a, v16bf b, v8f c) {
  return __builtin_amdgcn_wmma_f32_16x16x32_bf16(false, a, false, b,
                                                 (short)0, c, false, false);
}

// A/B operand fragment: 16 contiguous halves at p (two b128 loads).
__device__ __forceinline__ v16bf ld_frag1(const bf16_t* p) {
  const uint4 a = *(const uint4*)p;
  const uint4 b = *(const uint4*)(p + 8);
  v8u u;
  u[0] = a.x; u[1] = a.y; u[2] = a.z; u[3] = a.w;
  u[4] = b.x; u[5] = b.y; u[6] = b.z; u[7] = b.w;
  return __builtin_bit_cast(v16bf, u);
}

// A-layout fragment: elems 0..7 at p, elems 8..15 at p+16 (two b128 loads).
__device__ __forceinline__ v16bf ld_frag2(const bf16_t* p) {
  const uint4 a = *(const uint4*)p;
  const uint4 b = *(const uint4*)(p + 16);
  v8u u;
  u[0] = a.x; u[1] = a.y; u[2] = a.z; u[3] = a.w;
  u[4] = b.x; u[5] = b.y; u[6] = b.z; u[7] = b.w;
  return __builtin_bit_cast(v16bf, u);
}

// Async copy of 64 contiguous bytes global->LDS (4x ASYNC_TO_LDS_B128).
// ISA: INST_OFFSET is added to BOTH the LDS and the global address, so one
// LDS-dest VGPR + one 64-bit global address covers the whole chunk.
__device__ __forceinline__ void async_copy64(unsigned lds_off, const void* gsrc) {
  asm volatile(
      "global_load_async_to_lds_b128 %0, %1, off\n\t"
      "global_load_async_to_lds_b128 %0, %1, off offset:16\n\t"
      "global_load_async_to_lds_b128 %0, %1, off offset:32\n\t"
      "global_load_async_to_lds_b128 %0, %1, off offset:48"
      :
      : "v"(lds_off), "v"(gsrc)
      : "memory");
}

__device__ __forceinline__ void wait_async0() {
  asm volatile("s_wait_asynccnt 0" ::: "memory");
}

// generic LDS pointer -> LDS byte offset (low 32 bits of the address)
__device__ __forceinline__ unsigned lds_off32(const void* p) {
  return (unsigned)(unsigned long long)p;
}

// ---------------------------------------------------------------------------
// Generic GEMM: C[M,N] = A[M,K] @ W[K,N] + bias[N] (+ R[M,N] residual).
// Block: 256 threads (8 waves), tile 128x64, BK=32.
// Waves arranged 4x2, each wave computes 32x32 = 2x2 WMMA tiles.
// LDS: As[row][k], Bs[col][k] (B transposed; transpose done on the LDS-store
// side so global reads stay two b128 per thread from one address).
// ---------------------------------------------------------------------------
__global__ void __launch_bounds__(256)
k_gemm_bf16(const float* __restrict__ A, const float* __restrict__ W,
            const float* __restrict__ bias, const float* __restrict__ R,
            float* __restrict__ C, int M, int N, int K) {
  __shared__ bf16_t As[128][40];   // 128 rows x (32+8 pad) halves (80B stride)
  __shared__ bf16_t Bs[64][40];    //  64 cols x (32+8 pad) halves

  const int tid  = threadIdx.x;
  const int lane = tid & 31, wid = tid >> 5;
  const int wr = wid >> 1, wc = wid & 1;
  const int row0 = blockIdx.y * 128;
  const int col0 = blockIdx.x * 64;

  v8f acc[2][2];
  const v8f zero8 = {0.f, 0.f, 0.f, 0.f, 0.f, 0.f, 0.f, 0.f};
#pragma unroll
  for (int i = 0; i < 2; ++i)
#pragma unroll
    for (int j = 0; j < 2; ++j) acc[i][j] = zero8;

  const int ar = tid >> 1, ac = (tid & 1) * 16;   // A staging: 2 threads/row
  const int br = tid >> 3, bc = (tid & 7) * 8;    // B staging: 8 threads/row

  const int m16 = lane & 15;
  const int kbA = (lane >> 4) << 3;    // 0 or 8; unused (A frags use kbB)
  const int kbB = (lane >> 4) << 4;    // B frag K-base: 0 or 16

  const float* aP = A + (size_t)(row0 + ar) * K + ac;
  const float* wP = W + (size_t)br * N + col0 + bc;
  const size_t wStep = (size_t)32 * N;

  for (int k0 = 0; k0 < K; k0 += 32) {
    __syncthreads();
    {  // A tile: 16 contiguous fp32 -> two 16B bf16 stores
      const float4* a4 = (const float4*)(aP + k0);
      const float4 x0 = a4[0], x1 = a4[1], x2 = a4[2], x3 = a4[3];
      v8bf h0, h1;
      h0[0] = f2bf(x0.x); h0[1] = f2bf(x0.y); h0[2] = f2bf(x0.z); h0[3] = f2bf(x0.w);
      h0[4] = f2bf(x1.x); h0[5] = f2bf(x1.y); h0[6] = f2bf(x1.z); h0[7] = f2bf(x1.w);
      h1[0] = f2bf(x2.x); h1[1] = f2bf(x2.y); h1[2] = f2bf(x2.z); h1[3] = f2bf(x2.w);
      h1[4] = f2bf(x3.x); h1[5] = f2bf(x3.y); h1[6] = f2bf(x3.z); h1[7] = f2bf(x3.w);
      *(v8bf*)&As[ar][ac]     = h0;
      *(v8bf*)&As[ar][ac + 8] = h1;
    }
    {  // B tile: row-major b128 reads, transpose on the LDS-store side
      const float4* w4 = (const float4*)wP;
      const float4 y0 = w4[0], y1 = w4[1];
      v8bf hb;
      hb[0] = f2bf(y0.x); hb[1] = f2bf(y0.y); hb[2] = f2bf(y0.z); hb[3] = f2bf(y0.w);
      hb[4] = f2bf(y1.x); hb[5] = f2bf(y1.y); hb[6] = f2bf(y1.z); hb[7] = f2bf(y1.w);
#pragma unroll
      for (int i = 0; i < 8; ++i) Bs[bc + i][br] = hb[i];   // imm-offset b16 stores
    }
    if (k0 + 32 < K) {   // cache hints for next tiles (global_prefetch)
      __builtin_prefetch(aP + k0 + 32, 0, 0);
      __builtin_prefetch(wP + wStep, 0, 0);
    }
    wP += wStep;
    __syncthreads();

    v16bf af[2], bfr[2];
#pragma unroll
    for (int sm = 0; sm < 2; ++sm)
      af[sm] = ld_frag1(&As[wr * 32 + sm * 16 + m16][kbB]);
#pragma unroll
    for (int sn = 0; sn < 2; ++sn)
      bfr[sn] = ld_frag1(&Bs[wc * 32 + sn * 16 + m16][kbB]);
#pragma unroll
    for (int sm = 0; sm < 2; ++sm)
#pragma unroll
      for (int sn = 0; sn < 2; ++sn)
        acc[sm][sn] = wmma_bf16(af[sm], bfr[sn], acc[sm][sn]);
  }

  const int hi = (lane >> 4) << 3;     // C layout: lanes 16-31 -> rows +8
  const int colL = lane & 15;
#pragma unroll
  for (int sm = 0; sm < 2; ++sm)
#pragma unroll
    for (int sn = 0; sn < 2; ++sn)
#pragma unroll
      for (int j = 0; j < 8; ++j) {
        const int row = row0 + wr * 32 + sm * 16 + j + hi;
        const int col = col0 + wc * 32 + sn * 16 + colL;
        float v = acc[sm][sn][j] + bias[col];
        if (R) v += R[(size_t)row * N + col];
        C[(size_t)row * N + col] = v;
      }
}

// ---------------------------------------------------------------------------
// Fused flash attention over packed segments.
//   Qp,Kp : [nmat, 1024, 64] bf16 (Q pre-scaled by d^-0.5)
//   Vp    : [nmat, 64, 1024] bf16 (TRANSPOSED so PV fragments are contiguous)
//   Op    : [nmat, 1024, 64] fp32
// Causal over packed index. Block: 128 threads (4 waves), 64 q-rows/block;
// grid (1024/64, nmat). K/V tiles staged via async global->LDS (ASYNCcnt).
// ---------------------------------------------------------------------------
__global__ void __launch_bounds__(128)
k_flash(const bf16_t* __restrict__ Qp, const bf16_t* __restrict__ Kp,
        const bf16_t* __restrict__ Vp, float* __restrict__ Op) {
  __shared__ bf16_t Kt[64][72];      // [kpos][d]
  __shared__ bf16_t Vt[64][72];      // [d][kpos]  (transposed)
  __shared__ bf16_t Pl[4][16][72];   // per-wave P tile for C->A re-layout

  const int tid = threadIdx.x, lane = tid & 31, wave = tid >> 5;
  const int mat = blockIdx.y, qt = blockIdx.x;
  const bf16_t* Qg = Qp + (size_t)mat * SP * HDIM;
  const bf16_t* Kg = Kp + (size_t)mat * SP * HDIM;
  const bf16_t* Vg = Vp + (size_t)mat * SP * HDIM;   // [64][1024]
  float*        Og = Op + (size_t)mat * SP * HDIM;

  const int m16 = lane & 15;
  const int kbA = (lane >> 4) << 3;
  const int kbB = (lane >> 4) << 4;
  const int hi  = (lane >> 4) << 3;
  const v8f zero8 = {0.f, 0.f, 0.f, 0.f, 0.f, 0.f, 0.f, 0.f};

  // Q fragments for this wave's 16 rows (d split 0..31 / 32..63), b128 loads
  v16bf aq[2];
  {
    const int qrow = qt * 64 + wave * 16 + m16;
#pragma unroll
    for (int f = 0; f < 2; ++f)
      aq[f] = ld_frag1(Qg + (size_t)qrow * HDIM + f * 32 + kbB);
  }

  v8f o[4];
  float mrow[8], lrow[8];
#pragma unroll
  for (int t = 0; t < 4; ++t) o[t] = zero8;
#pragma unroll
  for (int j = 0; j < 8; ++j) { mrow[j] = -1e30f; lrow[j] = 0.f; }

  const int cr = tid >> 1;            // cooperative tile copy indices
  const int cc = (tid & 1) * 32;
  const unsigned kofs = lds_off32(&Kt[cr][cc]);
  const unsigned vofs = lds_off32(&Vt[cr][cc]);

  for (int kt = 0; kt <= qt; ++kt) {
    __syncthreads();
    // K tile: row kpos of Kg;  V tile: row d of transposed Vg.
    // 64B per thread per operand, direct to LDS (no VGPR staging).
    async_copy64(kofs, Kg + (size_t)(kt * 64 + cr) * HDIM + cc);
    async_copy64(vofs, Vg + (size_t)cr * SP + kt * 64 + cc);
    wait_async0();
    __syncthreads();

    // S = Q @ K^T  (16 x 64 logits per wave); B fragments contiguous in Kt
    v8f s[4];
#pragma unroll
    for (int t = 0; t < 4; ++t) {
      const v16bf b0 = ld_frag1(&Kt[t * 16 + m16][kbB]);
      const v16bf b1 = ld_frag1(&Kt[t * 16 + m16][32 + kbB]);
      s[t] = wmma_bf16(aq[0], b0, zero8);
      s[t] = wmma_bf16(aq[1], b1, s[t]);
    }

    if (kt == qt) {   // causal mask on the diagonal tile
#pragma unroll
      for (int t = 0; t < 4; ++t)
#pragma unroll
        for (int j = 0; j < 8; ++j)
          if (t * 16 + m16 > wave * 16 + j + hi) s[t][j] = -1e30f;
    }

    // online softmax (row j+hi lives on this lane's 16-lane half)
#pragma unroll
    for (int j = 0; j < 8; ++j) {
      float mx = fmaxf(fmaxf(s[0][j], s[1][j]), fmaxf(s[2][j], s[3][j]));
#pragma unroll
      for (int msk = 1; msk < 16; msk <<= 1)
        mx = fmaxf(mx, __shfl_xor(mx, msk, 32));
      const float mnew = fmaxf(mrow[j], mx);
      const float corr = __expf(mrow[j] - mnew);
      float rs = 0.f;
#pragma unroll
      for (int t = 0; t < 4; ++t) {
        const float p = __expf(s[t][j] - mnew);
        s[t][j] = p;
        rs += p;
      }
#pragma unroll
      for (int msk = 1; msk < 16; msk <<= 1)
        rs += __shfl_xor(rs, msk, 32);
      lrow[j] = lrow[j] * corr + rs;
      mrow[j] = mnew;
#pragma unroll
      for (int t = 0; t < 4; ++t) o[t][j] *= corr;
    }

    // P (C-layout) -> LDS -> A-layout fragments
#pragma unroll
    for (int t = 0; t < 4; ++t)
#pragma unroll
      for (int j = 0; j < 8; ++j)
        Pl[wave][j + hi][t * 16 + m16] = f2bf(s[t][j]);

    v16bf ap[2];
#pragma unroll
    for (int f = 0; f < 2; ++f)
      ap[f] = ld_frag1(&Pl[wave][m16][f * 32 + kbB]);

    // O += P @ V ; B fragments contiguous in transposed Vt
#pragma unroll
    for (int t = 0; t < 4; ++t) {
      const v16bf bv0 = ld_frag1(&Vt[t * 16 + m16][kbB]);
      const v16bf bv1 = ld_frag1(&Vt[t * 16 + m16][32 + kbB]);
      o[t] = wmma_bf16(ap[0], bv0, o[t]);
      o[t] = wmma_bf16(ap[1], bv1, o[t]);
    }
  }

#pragma unroll
  for (int t = 0; t < 4; ++t)
#pragma unroll
    for (int j = 0; j < 8; ++j) {
      const int row = qt * 64 + wave * 16 + j + hi;
      Og[(size_t)row * HDIM + t * 16 + m16] = o[t][j] / lrow[j];
    }
}

// ---------------------------------------------------------------------------
// Elementwise / reduction kernels
// ---------------------------------------------------------------------------
__device__ __forceinline__ float block_sum(float v, float* red) {
#pragma unroll
  for (int m = 16; m >= 1; m >>= 1) v += __shfl_xor(v, m, 32);
  __syncthreads();
  if ((threadIdx.x & 31) == 0) red[threadIdx.x >> 5] = v;
  __syncthreads();
  float t = (threadIdx.x < 8) ? red[threadIdx.x] : 0.f;
#pragma unroll
  for (int m = 4; m >= 1; m >>= 1) t += __shfl_xor(t, m, 32);
  if (threadIdx.x == 0) red[0] = t;
  __syncthreads();
  return red[0];
}

__global__ void __launch_bounds__(256)
k_ln(const float* __restrict__ X, const float* __restrict__ g,
     const float* __restrict__ b, float* __restrict__ Y, int dim, int relu_in) {
  __shared__ float red[8];
  const float* xr = X + (size_t)blockIdx.x * dim;
  float* yr = Y + (size_t)blockIdx.x * dim;
  const int nv = dim >> 8;   // dim / 256: 3 (768) or 12 (3072)
  float vals[12];
  float s = 0.f;
  for (int i = 0; i < nv; ++i) {
    float v = xr[threadIdx.x + (i << 8)];
    if (relu_in) v = fmaxf(v, 0.f);
    vals[i] = v; s += v;
  }
  const float mu = block_sum(s, red) / (float)dim;
  float s2 = 0.f;
  for (int i = 0; i < nv; ++i) { const float d = vals[i] - mu; s2 += d * d; }
  const float var = block_sum(s2, red) / (float)dim;
  const float rstd = rsqrtf(var + 1e-5f);
  for (int i = 0; i < nv; ++i) {
    const int c = threadIdx.x + (i << 8);
    yr[c] = (vals[i] - mu) * rstd * g[c] + b[c];
  }
}

// h = e + xpos(e) with e = emb[x];  grid SEQN, block 128 (3 dim-pairs/thread)
__global__ void __launch_bounds__(128)
k_embed(const int* __restrict__ x, const float* __restrict__ emb,
        float* __restrict__ h) {
  const int pos = blockIdx.x;
  const int tok = x[pos];
  const float* e = emb + (size_t)tok * DMODEL;
  float* hr = h + (size_t)pos * DMODEL;
  const float power = ((float)pos - 2048.0f) * (1.0f / 512.0f);
  for (int t = 0; t < 3; ++t) {
    const int j = threadIdx.x + t * 128;                       // 0..383
    const float sv = (2.0f * j + 0.4f * 768.0f) / (1.4f * 768.0f);
    // powf -> exp/log (transcendental HW ops instead of ocml pow blob)
    const float scale = __expf(power * __logf(sv));
    const float inv_freq = __expf((float)j * (-9.210340371976184f / 384.0f));
    const float ang = (float)pos * inv_freq;
    const float sn = __sinf(ang) * scale;
    const float cs = __cosf(ang) * scale;
    const float e0 = e[2 * j], e1 = e[2 * j + 1];
    hr[2 * j]     = e0 + (e0 * cs - e1 * sn);
    hr[2 * j + 1] = e1 + (e1 * cs + e0 * sn);
  }
}

__global__ void k_zero(float* __restrict__ p, int n) {
  const int i = blockIdx.x * blockDim.x + threadIdx.x;
  if (i < n) p[i] = 0.f;
}

// gather 4096x768 fp32 -> packed [nmat,1024,64] bf16 (Q pre-scaled)
__global__ void k_pack(const float* __restrict__ X, bf16_t* __restrict__ P,
                       int s, int r, int off, int hmin, float mul) {
  const int idx = blockIdx.x * blockDim.x + threadIdx.x;
  const int mat = idx >> 16;
  const int p   = (idx >> 6) & 1023;
  const int dd  = idx & 63;
  const int seq  = (mat >> 2) * s + off + p * r;
  const int head = hmin + (mat & 3);
  P[idx] = f2bf(X[(size_t)seq * DMODEL + head * HDIM + dd] * mul);
}

// gather -> packed TRANSPOSED [nmat,64,1024] bf16 (for V: contiguous PV frags)
__global__ void k_packT(const float* __restrict__ X, bf16_t* __restrict__ P,
                        int s, int r, int off, int hmin) {
  const int idx = blockIdx.x * blockDim.x + threadIdx.x;
  const int mat = idx >> 16;
  const int dd  = (idx >> 10) & 63;
  const int p   = idx & 1023;
  const int seq  = (mat >> 2) * s + off + p * r;
  const int head = hmin + (mat & 3);
  P[idx] = f2bf(X[(size_t)seq * DMODEL + head * HDIM + dd]);
}

// scatter packed fp32 back into [4096,768] accumulator; slices are disjoint,
// average over the 3 configs folded in as 1/3.
__global__ void k_scatter(const float* __restrict__ Op, float* __restrict__ Out,
                          int s, int r, int off, int hmin) {
  const int idx = blockIdx.x * blockDim.x + threadIdx.x;
  const int mat = idx >> 16;
  const int p   = (idx >> 6) & 1023;
  const int dd  = idx & 63;
  const int seq  = (mat >> 2) * s + off + p * r;
  const int head = hmin + (mat & 3);
  Out[(size_t)seq * DMODEL + head * HDIM + dd] = Op[idx] * (1.0f / 3.0f);
}

// ---------------------------------------------------------------------------
// Host orchestration.
// Input flattening (setup_inputs insertion order), 110 leaf arrays:
//   0:x  1:emb
//   enc L (L=0,1) @ base=2+20L: norm1 g,b | attn(qw,qb,kw,kb,vw,vb,lng,lnb,ow,ob)
//                               | norm2 g,b | lin1 w,b | norm3 g,b | lin2 w,b
//   dec L (L=0,1) @ base=42+32L: norm1 g,b | self_attn(10) | norm2 g,b
//                               | cross_attn(10) | norm3 g,b | lin1 w,b
//                               | norm4 g,b | lin2 w,b
//   106,107: final_norm g,b    108,109: out w,b
// ---------------------------------------------------------------------------
extern "C" void kernel_launch(void* const* d_in, const int* in_sizes, int n_in,
                              void* d_out, int out_size, void* d_ws, size_t ws_size,
                              hipStream_t stream) {
  (void)in_sizes; (void)n_in; (void)out_size; (void)ws_size;
  auto F = [&](int i) { return (const float*)d_in[i]; };
  const int*   xtok = (const int*)d_in[0];
  const float* emb  = F(1);

  char* w = (char*)d_ws;
  auto carve = [&](size_t bytes) {
    void* p = (void*)w;
    w += (bytes + 255) & ~(size_t)255;
    return p;
  };
  float* mem   = (float*)carve((size_t)SEQN * DMODEL * 4);
  float* tgt   = (float*)carve((size_t)SEQN * DMODEL * 4);
  float* ybuf  = (float*)carve((size_t)SEQN * DMODEL * 4);
  float* qbuf  = (float*)carve((size_t)SEQN * DMODEL * 4);
  float* kbuf  = (float*)carve((size_t)SEQN * DMODEL * 4);
  float* vbuf  = (float*)carve((size_t)SEQN * DMODEL * 4);
  float* attnb = (float*)carve((size_t)SEQN * DMODEL * 4);
  float* ffn   = (float*)carve((size_t)SEQN * DFF * 4);
  bf16_t* qp   = (bf16_t*)carve((size_t)16 * SP * HDIM * 2);
  bf16_t* kp   = (bf16_t*)carve((size_t)16 * SP * HDIM * 2);
  bf16_t* vp   = (bf16_t*)carve((size_t)16 * SP * HDIM * 2);
  float*  opb  = (float*)carve((size_t)16 * SP * HDIM * 4);

  auto gemm = [&](const float* A, const float* Wm, const float* bias,
                  const float* R, float* Cc, int M, int N, int K) {
    dim3 grid(N / 64, M / 128);
    k_gemm_bf16<<<grid, 256, 0, stream>>>(A, Wm, bias, R, Cc, M, N, K);
  };
  auto lnorm = [&](const float* X, const float* gg, const float* bb, float* Y,
                   int dim, int relu) {
    k_ln<<<SEQN, 256, 0, stream>>>(X, gg, bb, Y, dim, relu);
  };
  auto attention = [&](const float* qin, const float* kvin, int pb, float* dest) {
    gemm(qin,  F(pb + 0), F(pb + 1), nullptr, qbuf, SEQN, DMODEL, DMODEL);
    gemm(kvin, F(pb + 2), F(pb + 3), nullptr, kbuf, SEQN, DMODEL, DMODEL);
    gemm(kvin, F(pb + 4), F(pb + 5), nullptr, vbuf, SEQN, DMODEL, DMODEL);
    k_zero<<<(SEQN * DMODEL) / 256, 256, 0, stream>>>(attnb, SEQN * DMODEL);
    const int Sseg[3] = {1024, 2048, 4096};
    const int Rdil[3] = {1, 2, 4};
    for (int i = 0; i < 3; ++i) {
      const int s = Sseg[i], r = Rdil[i];
      const int off = i % r;
      const int hmin = i * 4;
      const int nmat = (SEQN / s) * 4;
      const int nel = nmat * SP * HDIM;
      k_pack <<<nel / 256, 256, 0, stream>>>(qbuf, qp, s, r, off, hmin, 0.125f);
      k_pack <<<nel / 256, 256, 0, stream>>>(kbuf, kp, s, r, off, hmin, 1.0f);
      k_packT<<<nel / 256, 256, 0, stream>>>(vbuf, vp, s, r, off, hmin);
      k_flash<<<dim3(SP / 64, nmat), 128, 0, stream>>>(qp, kp, vp, opb);
      k_scatter<<<nel / 256, 256, 0, stream>>>(opb, attnb, s, r, off, hmin);
    }
    lnorm(attnb, F(pb + 6), F(pb + 7), qbuf, DMODEL, 0);                // inner LN
    gemm(qbuf, F(pb + 8), F(pb + 9), dest, dest, SEQN, DMODEL, DMODEL); // o-proj + residual
  };

  // embedding + xPos; tgt starts equal to mem
  k_embed<<<SEQN, 128, 0, stream>>>(xtok, emb, mem);
  hipMemcpyAsync(tgt, mem, (size_t)SEQN * DMODEL * 4,
                 hipMemcpyDeviceToDevice, stream);

  // encoder stack
  for (int L = 0; L < 2; ++L) {
    const int base = 2 + L * 20;
    lnorm(mem, F(base + 0), F(base + 1), ybuf, DMODEL, 0);
    attention(ybuf, ybuf, base + 2, mem);
    lnorm(mem, F(base + 12), F(base + 13), ybuf, DMODEL, 0);
    gemm(ybuf, F(base + 14), F(base + 15), nullptr, ffn, SEQN, DFF, DMODEL);
    lnorm(ffn, F(base + 16), F(base + 17), ffn, DFF, 1);              // LN(relu(.))
    gemm(ffn, F(base + 18), F(base + 19), mem, mem, SEQN, DMODEL, DFF);
  }
  // decoder stack
  for (int L = 0; L < 2; ++L) {
    const int base = 42 + L * 32;
    lnorm(tgt, F(base + 0), F(base + 1), ybuf, DMODEL, 0);
    attention(ybuf, ybuf, base + 2, tgt);                             // self
    lnorm(tgt, F(base + 12), F(base + 13), ybuf, DMODEL, 0);
    attention(ybuf, mem, base + 14, tgt);                             // cross
    lnorm(tgt, F(base + 24), F(base + 25), ybuf, DMODEL, 0);
    gemm(ybuf, F(base + 26), F(base + 27), nullptr, ffn, SEQN, DFF, DMODEL);
    lnorm(ffn, F(base + 28), F(base + 29), ffn, DFF, 1);
    gemm(ffn, F(base + 30), F(base + 31), tgt, tgt, SEQN, DMODEL, DFF);
  }

  // final norm + vocab head -> d_out (4096 x 16000 fp32)
  lnorm(tgt, F(106), F(107), ybuf, DMODEL, 0);
  gemm(ybuf, F(108), F(109), nullptr, (float*)d_out, SEQN, VOCABN, DMODEL);
}
